// HybridGraphClassifier_56092272886049
// MI455X (gfx1250) — matrix operand order repaired and checked by reading, not compiled, not run on hardware
//
#include <hip/hip_runtime.h>
#include <hip/hip_bf16.h>
#include <stdint.h>

// ---------------------------------------------------------------------------
// Types for CDNA5 WMMA (wave32): v_wmma_f32_16x16x32_bf16
// ---------------------------------------------------------------------------
typedef __attribute__((ext_vector_type(16))) __bf16 v16bf;
typedef __attribute__((ext_vector_type(8)))  float  v8f;

union AFrag { v16bf v; unsigned int u[8]; };

#define WMMA_BF16(a, b, c) \
  __builtin_amdgcn_wmma_f32_16x16x32_bf16(false, (a), false, (b), (short)0, (c), false, false)

__device__ __forceinline__ unsigned short cvt_bf16(float f) {
  unsigned int u = __float_as_uint(f);
  u += 0x7FFFu + ((u >> 16) & 1u);          // round-to-nearest-even
  return (unsigned short)(u >> 16);
}
__device__ __forceinline__ unsigned int pack_bf16x2(float lo, float hi) {
  return (unsigned int)cvt_bf16(lo) | ((unsigned int)cvt_bf16(hi) << 16);
}
__device__ __forceinline__ float bf16_to_f32(unsigned short h) {
  return __uint_as_float(((unsigned int)h) << 16);
}
__device__ __forceinline__ void atomAddF(float* p, float v) {
  unsafeAtomicAdd(p, v);                    // -> global_atomic_add_f32
}

// ---------------------------------------------------------------------------
// GEMM1: H1_bf16 = bf16(x_f32 @ W1T^T + b1), K=768, Nout=768
// one wave (32 threads) computes a 16x64 output tile
// ---------------------------------------------------------------------------
__global__ void gemm1_kernel(const float* __restrict__ X,
                             const unsigned short* __restrict__ W1T,  // [768][768] bf16 (Nout-major)
                             const float* __restrict__ b1,
                             unsigned short* __restrict__ H1,         // [M][768] bf16
                             int M) {
  const int K = 768, Nout = 768;
  const int lane = threadIdx.x & 31;
  const int ln15 = lane & 15;
  const int hiL  = lane >> 4;
  const int m0 = blockIdx.x * 16;
  const int n0 = blockIdx.y * 64;
  int rowA = m0 + ln15; if (rowA >= M) rowA = M - 1;
  const float* arow = X + (size_t)rowA * K;
  const int hk = hiL ? 16 : 0;   // B-frag K base per lane half
  const int ak = hiL ? 8  : 0;   // unused; A-frag also uses hk (k = 16h + s)
  v8f acc0 = {}, acc1 = {}, acc2 = {}, acc3 = {};
  for (int k0 = 0; k0 < K; k0 += 32) {
    AFrag a;
#pragma unroll
    for (int j = 0; j < 8; ++j) {
      int kk = hk + (j << 1);
      float2 f = *(const float2*)(arow + k0 + kk);
      a.u[j] = pack_bf16x2(f.x, f.y);
    }
    AFrag b0, b1f, b2, b3;
    const unsigned short* bp0 = W1T + (size_t)(n0 +  0 + ln15) * K + k0 + hk;
    const unsigned short* bp1 = W1T + (size_t)(n0 + 16 + ln15) * K + k0 + hk;
    const unsigned short* bp2 = W1T + (size_t)(n0 + 32 + ln15) * K + k0 + hk;
    const unsigned short* bp3 = W1T + (size_t)(n0 + 48 + ln15) * K + k0 + hk;
#pragma unroll
    for (int i = 0; i < 8; ++i) {
      b0.u[i] = *(const unsigned int*)(bp0 + 2 * i);
      b1f.u[i] = *(const unsigned int*)(bp1 + 2 * i);
      b2.u[i] = *(const unsigned int*)(bp2 + 2 * i);
      b3.u[i] = *(const unsigned int*)(bp3 + 2 * i);
    }
    acc0 = WMMA_BF16(a.v, b0.v, acc0);
    acc1 = WMMA_BF16(a.v, b1f.v, acc1);
    acc2 = WMMA_BF16(a.v, b2.v, acc2);
    acc3 = WMMA_BF16(a.v, b3.v, acc3);
  }
  float bv0 = b1[n0 + ln15], bv1 = b1[n0 + ln15 + 16];
  float bv2 = b1[n0 + ln15 + 32], bv3 = b1[n0 + ln15 + 48];
#pragma unroll
  for (int r = 0; r < 8; ++r) {
    int row = m0 + r + (hiL ? 8 : 0);
    if (row < M) {
      unsigned short* crow = H1 + (size_t)row * Nout + n0 + ln15;
      crow[0]  = cvt_bf16(acc0[r] + bv0);
      crow[16] = cvt_bf16(acc1[r] + bv1);
      crow[32] = cvt_bf16(acc2[r] + bv2);
      crow[48] = cvt_bf16(acc3[r] + bv3);
    }
  }
}

// ---------------------------------------------------------------------------
// Generic bf16 GEMM: C_f32 = A_bf16[M,K] @ Bt_bf16[Nout,K]^T (no bias)
// ---------------------------------------------------------------------------
__global__ void gemm_bf16_kernel(const unsigned short* __restrict__ A,
                                 const unsigned short* __restrict__ Bt,
                                 float* __restrict__ C,
                                 int M, int K, int Nout) {
  const int lane = threadIdx.x & 31;
  const int ln15 = lane & 15;
  const int hiL  = lane >> 4;
  const int m0 = blockIdx.x * 16;
  const int n0 = blockIdx.y * 64;
  int rowA = m0 + ln15; if (rowA >= M) rowA = M - 1;
  const unsigned short* arow = A + (size_t)rowA * K;
  const int hk = hiL ? 16 : 0;
  const int ak = hiL ? 8  : 0;
  v8f acc0 = {}, acc1 = {}, acc2 = {}, acc3 = {};
  for (int k0 = 0; k0 < K; k0 += 32) {
    AFrag a;
#pragma unroll
    for (int j = 0; j < 8; ++j) {
      int kk = hk + (j << 1);
      a.u[j] = *(const unsigned int*)(arow + k0 + kk);
    }
    AFrag b0, b1f, b2, b3;
    const unsigned short* bp0 = Bt + (size_t)(n0 +  0 + ln15) * K + k0 + hk;
    const unsigned short* bp1 = Bt + (size_t)(n0 + 16 + ln15) * K + k0 + hk;
    const unsigned short* bp2 = Bt + (size_t)(n0 + 32 + ln15) * K + k0 + hk;
    const unsigned short* bp3 = Bt + (size_t)(n0 + 48 + ln15) * K + k0 + hk;
#pragma unroll
    for (int i = 0; i < 8; ++i) {
      b0.u[i] = *(const unsigned int*)(bp0 + 2 * i);
      b1f.u[i] = *(const unsigned int*)(bp1 + 2 * i);
      b2.u[i] = *(const unsigned int*)(bp2 + 2 * i);
      b3.u[i] = *(const unsigned int*)(bp3 + 2 * i);
    }
    acc0 = WMMA_BF16(a.v, b0.v, acc0);
    acc1 = WMMA_BF16(a.v, b1f.v, acc1);
    acc2 = WMMA_BF16(a.v, b2.v, acc2);
    acc3 = WMMA_BF16(a.v, b3.v, acc3);
  }
#pragma unroll
  for (int r = 0; r < 8; ++r) {
    int row = m0 + r + (hiL ? 8 : 0);
    if (row < M) {
      float* crow = C + (size_t)row * Nout + n0 + ln15;
      crow[0]  = acc0[r];
      crow[16] = acc1[r];
      crow[32] = acc2[r];
      crow[48] = acc3[r];
    }
  }
}

// ---------------------------------------------------------------------------
// Small helper kernels
// ---------------------------------------------------------------------------
__global__ void transpose_to_bf16(const float* __restrict__ W,       // [K][Nout] f32
                                  unsigned short* __restrict__ Bt,   // [Nout][K] bf16
                                  int K, int Nout) {
  int idx = blockIdx.x * 256 + threadIdx.x;
  if (idx >= K * Nout) return;
  int k = idx / Nout, n = idx - k * Nout;
  Bt[(size_t)n * K + k] = cvt_bf16(W[idx]);
}

__global__ void fill_f32(float* __restrict__ p, float v, long long n) {
  long long i = (long long)blockIdx.x * 256 + threadIdx.x;
  if (i < n) p[i] = v;
}

__global__ void deg_edges(const int* __restrict__ dst, float* __restrict__ deg, int E) {
  int e = blockIdx.x * 256 + threadIdx.x;
  if (e < E) atomAddF(&deg[dst[e]], 1.0f);
}

__global__ void dinv_kernel(float* __restrict__ d, int N) {
  int n = blockIdx.x * 256 + threadIdx.x;
  if (n < N) d[n] = rsqrtf(d[n]);
}

__global__ void a2_kernel(const float* __restrict__ b2, const float* __restrict__ attn_w,
                          const float* __restrict__ attn_b, float* __restrict__ a2out) {
  __shared__ float red[256];
  float s = 0.f;
  for (int k = threadIdx.x; k < 768; k += 256) s += b2[k] * attn_w[k];
  red[threadIdx.x] = s; __syncthreads();
  for (int off = 128; off > 0; off >>= 1) {
    if ((int)threadIdx.x < off) red[threadIdx.x] += red[threadIdx.x + off];
    __syncthreads();
  }
  if (threadIdx.x == 0) a2out[0] = red[0] + attn_b[0];
}

// per-node attention softmax fusion: fused = w0*h1 + w1*b2 (h2 == fusion_b2)
__global__ void fusion_epilogue(const unsigned short* __restrict__ h1,
                                const float* __restrict__ attn_w, const float* __restrict__ attn_b,
                                const float* __restrict__ b2, const float* __restrict__ a2s,
                                unsigned short* __restrict__ fused) {
  __shared__ float red[256];
  const int n = blockIdx.x;
  const unsigned short* row = h1 + (size_t)n * 768;
  float s = 0.f;
  for (int c = threadIdx.x; c < 768; c += 256) s += bf16_to_f32(row[c]) * attn_w[c];
  red[threadIdx.x] = s; __syncthreads();
  for (int off = 128; off > 0; off >>= 1) {
    if ((int)threadIdx.x < off) red[threadIdx.x] += red[threadIdx.x + off];
    __syncthreads();
  }
  float a1 = red[0] + attn_b[0];
  float a2 = a2s[0];
  float mx = fmaxf(a1, a2);
  float e1 = __expf(a1 - mx), e2 = __expf(a2 - mx);
  float inv = 1.f / (e1 + e2);
  float w0 = e1 * inv, w1 = e2 * inv;
  for (int c = threadIdx.x; c < 768; c += 256) {
    float v = w0 * bf16_to_f32(row[c]) + w1 * b2[c];
    fused[(size_t)n * 768 + c] = cvt_bf16(v);
  }
}

// one wave per edge; lane covers 8 of 256 channels (coalesced), f32 atomics
__global__ void scatter_kernel(const float* __restrict__ t, const int* __restrict__ src,
                               const int* __restrict__ dst, const float* __restrict__ dinv,
                               float* __restrict__ agg, int E) {
  int wid = (int)((blockIdx.x * (unsigned)blockDim.x + threadIdx.x) >> 5);
  int lane = threadIdx.x & 31;
  if (wid >= E) return;
  int s = src[wid], d = dst[wid];
  float norm = dinv[s] * dinv[d];
  const float* trow = t + (size_t)s * 256;
  float* arow = agg + (size_t)d * 256;
#pragma unroll
  for (int i = 0; i < 8; ++i) {
    int c = lane + i * 32;
    atomAddF(&arow[c], trow[c] * norm);
  }
}

// v = relu(agg + t*dinv^2 + bias); writes bf16 (layer1) or f32 (layer2)
__global__ void finish_kernel(const float* __restrict__ t, const float* __restrict__ agg,
                              const float* __restrict__ dinv, const float* __restrict__ bias,
                              unsigned short* __restrict__ obf, float* __restrict__ of32, int N) {
  long long idx = (long long)blockIdx.x * 256 + threadIdx.x;
  if (idx >= (long long)N * 256) return;
  int n = (int)(idx >> 8), c = (int)(idx & 255);
  float di = dinv[n];
  float v = agg[idx] + t[idx] * di * di + bias[c];
  v = fmaxf(v, 0.f);
  if (obf) obf[idx] = cvt_bf16(v);
  else     of32[idx] = v;
}

__global__ void count_kernel(const int* __restrict__ batch, float* __restrict__ counts, int N) {
  int n = blockIdx.x * 256 + threadIdx.x;
  if (n < N) atomAddF(&counts[batch[n]], 1.0f);
}

// batch is sorted -> run-compacted per-graph partial sums, few atomics
__global__ void pool_kernel(const float* __restrict__ h, const int* __restrict__ batch,
                            float* __restrict__ pools, int N) {
  int n0 = blockIdx.x * 64;
  if (n0 >= N) return;
  int c = threadIdx.x;                      // 256 channels
  int nend = n0 + 64; if (nend > N) nend = N;
  int gcur = batch[n0];
  float s = 0.f;
  for (int n = n0; n < nend; ++n) {
    int g = batch[n];
    if (g != gcur) { atomAddF(&pools[(size_t)gcur * 256 + c], s); s = 0.f; gcur = g; }
    s += h[(size_t)n * 256 + c];
  }
  atomAddF(&pools[(size_t)gcur * 256 + c], s);
}

__global__ void cls1_kernel(const float* __restrict__ pools, const float* __restrict__ counts,
                            const float* __restrict__ w1, const float* __restrict__ b1,
                            float* __restrict__ z) {
  int g = blockIdx.x, j = threadIdx.x;      // 128 threads
  float inv = 1.0f / fmaxf(counts[g], 1.0f);
  float s = 0.f;
  for (int c = 0; c < 256; ++c) s += pools[(size_t)g * 256 + c] * w1[c * 128 + j];
  z[g * 128 + j] = fmaxf(s * inv + b1[j], 0.f);
}

__global__ void cls2_kernel(const float* __restrict__ z, const float* __restrict__ w2,
                            const float* __restrict__ b2, float* __restrict__ out) {
  int g = blockIdx.x, q = threadIdx.x;
  if (q >= 5) return;
  float s = 0.f;
  for (int j = 0; j < 128; ++j) s += z[g * 128 + j] * w2[j * 5 + q];
  out[g * 5 + q] = s + b2[q];
}

// ---------------------------------------------------------------------------
// Launch
// ---------------------------------------------------------------------------
extern "C" void kernel_launch(void* const* d_in, const int* in_sizes, int n_in,
                              void* d_out, int out_size, void* d_ws, size_t ws_size,
                              hipStream_t stream) {
  const float* x         = (const float*)d_in[0];
  const int*   ei        = (const int*)d_in[1];
  const int*   batch     = (const int*)d_in[2];
  const float* fusion_w1 = (const float*)d_in[3];
  const float* fusion_b1 = (const float*)d_in[4];
  /* d_in[5] = fusion_w2 (unused: node2vec input is zeros) */
  const float* fusion_b2 = (const float*)d_in[6];
  const float* attn_w    = (const float*)d_in[7];
  const float* attn_b    = (const float*)d_in[8];
  const float* conv1_w   = (const float*)d_in[9];
  const float* conv1_b   = (const float*)d_in[10];
  const float* conv2_w   = (const float*)d_in[11];
  const float* conv2_b   = (const float*)d_in[12];
  const float* cls_w1    = (const float*)d_in[13];
  const float* cls_b1    = (const float*)d_in[14];
  const float* cls_w2    = (const float*)d_in[15];
  const float* cls_b2    = (const float*)d_in[16];

  const int N = in_sizes[0] / 768;
  const int E = in_sizes[1] / 2;
  const int G = out_size / 5;
  const int* src = ei;
  const int* dst = ei + E;

  char* ws = (char*)d_ws;
  size_t off = 0;
  auto take = [&](size_t bytes) -> char* {
    char* p = ws + off;
    off = (off + bytes + 255) & ~(size_t)255;
    return p;
  };
  unsigned short* W1T    = (unsigned short*)take((size_t)768 * 768 * 2);
  unsigned short* C1T    = (unsigned short*)take((size_t)768 * 256 * 2);
  unsigned short* C2T    = (unsigned short*)take((size_t)256 * 256 * 2);
  float*          dinv   = (float*)take((size_t)N * 4);
  float*          a2s    = (float*)take(256);
  float*          pools  = (float*)take((size_t)G * 256 * 4);
  float*          counts = (float*)take((size_t)G * 4);
  float*          zh     = (float*)take((size_t)G * 128 * 4);
  unsigned short* H1     = (unsigned short*)take((size_t)N * 768 * 2);  // h1 bf16
  unsigned short* FUSED  = (unsigned short*)take((size_t)N * 768 * 2);  // fused bf16
  float*          T      = (float*)take((size_t)N * 256 * 4);           // GEMM out
  float*          AGG    = (float*)take((size_t)N * 256 * 4);           // scatter acc
  unsigned short* HACT   = (unsigned short*)take((size_t)N * 256 * 2);  // layer1 act bf16
  float*          H2     = (float*)H1;  // reuse: h1 dead after fusion epilogue

  const dim3 b256(256);
  const long long nAgg = (long long)N * 256;
  const unsigned aggBlocks = (unsigned)((nAgg + 255) / 256);

  // weight prep (transpose to [Nout][K] bf16 so B fragments load packed K-pairs)
  transpose_to_bf16<<<(768 * 768 + 255) / 256, b256, 0, stream>>>(fusion_w1, W1T, 768, 768);
  transpose_to_bf16<<<(768 * 256 + 255) / 256, b256, 0, stream>>>(conv1_w, C1T, 768, 256);
  transpose_to_bf16<<<(256 * 256 + 255) / 256, b256, 0, stream>>>(conv2_w, C2T, 256, 256);

  // degrees: deg = 1 + indegree; then dinv = rsqrt(deg)
  fill_f32<<<(N + 255) / 256, b256, 0, stream>>>(dinv, 1.0f, N);
  deg_edges<<<(E + 255) / 256, b256, 0, stream>>>(dst, dinv, E);
  dinv_kernel<<<(N + 255) / 256, b256, 0, stream>>>(dinv, N);

  // constant attention logit for the zero node2vec branch
  a2_kernel<<<1, b256, 0, stream>>>(fusion_b2, attn_w, attn_b, a2s);

  // zero accumulators (ws is poisoned, and must be re-zeroed every call)
  fill_f32<<<aggBlocks, b256, 0, stream>>>(AGG, 0.0f, nAgg);
  fill_f32<<<(G * 256 + 255) / 256, b256, 0, stream>>>(pools, 0.0f, (long long)G * 256);
  fill_f32<<<1, b256, 0, stream>>>(counts, 0.0f, G);

  // GEMM1 (WMMA bf16) + attention fusion
  dim3 g1((N + 15) / 16, 768 / 64);
  gemm1_kernel<<<g1, dim3(32), 0, stream>>>(x, W1T, fusion_b1, H1, N);
  fusion_epilogue<<<N, b256, 0, stream>>>(H1, attn_w, attn_b, fusion_b2, a2s, FUSED);

  // GCN layer 1
  dim3 g2((N + 15) / 16, 256 / 64);
  gemm_bf16_kernel<<<g2, dim3(32), 0, stream>>>(FUSED, C1T, T, N, 768, 256);
  scatter_kernel<<<(unsigned)(((long long)E * 32 + 255) / 256), b256, 0, stream>>>(T, src, dst, dinv, AGG, E);
  finish_kernel<<<aggBlocks, b256, 0, stream>>>(T, AGG, dinv, conv1_b, HACT, nullptr, N);

  // GCN layer 2
  fill_f32<<<aggBlocks, b256, 0, stream>>>(AGG, 0.0f, nAgg);
  gemm_bf16_kernel<<<g2, dim3(32), 0, stream>>>(HACT, C2T, T, N, 256, 256);
  scatter_kernel<<<(unsigned)(((long long)E * 32 + 255) / 256), b256, 0, stream>>>(T, src, dst, dinv, AGG, E);
  finish_kernel<<<aggBlocks, b256, 0, stream>>>(T, AGG, dinv, conv2_b, nullptr, H2, N);

  // global mean pool + classifier
  count_kernel<<<(N + 255) / 256, b256, 0, stream>>>(batch, counts, N);
  pool_kernel<<<(N + 63) / 64, b256, 0, stream>>>(H2, batch, pools, N);
  cls1_kernel<<<G, dim3(128), 0, stream>>>(pools, counts, cls_w1, cls_b1, zh);
  cls2_kernel<<<G, dim3(32), 0, stream>>>(zh, cls_w2, cls_b2, (float*)d_out);
}
